// LocalRNN_61641370632256
// MI455X (gfx1250) — compile-verified, matrix operand order checked by reading, not run on hardware
//
#include <hip/hip_runtime.h>

typedef __attribute__((ext_vector_type(16))) _Float16 v16h;
typedef __attribute__((ext_vector_type(8)))  float    v8f;
typedef __attribute__((ext_vector_type(4)))  float    f4;

#define Bb 8
#define Ll 2048
#define Dd 768
#define N3 2304            // 3*D
#define KW 8               // window size
#define MT 32              // positions per workgroup
#define XH_PITCH 776       // halves per row (768 + 8 pad -> conflict-free b128)
#define XH_ROWS  48        // phase1 x rows (3 M-blocks); phase2 uses rows 0..31 for h16
#define GX_PITCH 2312      // halves per row (2304 + 8 pad)
#define GX_ROWS  40        // rows 0..38 needed (m<=31, j<=7)
#define XH_BYTES (XH_ROWS * XH_PITCH * 2)      // 74,496
#define GX_BYTES (GX_ROWS * GX_PITCH * 2)      // 184,960
#define LDS_BYTES (XH_BYTES + GX_BYTES)        // 259,456

union Frag { v16h v; f4 q[2]; };

__device__ __forceinline__ float sigf(float x) { return 1.0f / (1.0f + __expf(-x)); }
__device__ __forceinline__ float tanh_fast(float x) {
    float ax = fabsf(x);
    float e  = __expf(-2.0f * ax);
    float t  = (1.0f - e) / (1.0f + e);
    return copysignf(t, x);
}

// Convert weights to transposed f16: wT[n][k] = (f16)w[k][n]
__global__ void prep_weights(const float* __restrict__ wih, const float* __restrict__ whh,
                             _Float16* __restrict__ wihT, _Float16* __restrict__ whhT) {
    int idx = blockIdx.x * blockDim.x + threadIdx.x;
    if (idx >= Dd * N3) return;
    int k = idx / N3, n = idx - k * N3;
    wihT[(size_t)n * Dd + k] = (_Float16)wih[idx];
    whhT[(size_t)n * Dd + k] = (_Float16)whh[idx];
}

__global__ __launch_bounds__(256) void localrnn_kernel(
    const float* __restrict__ x, const _Float16* __restrict__ wihT,
    const _Float16* __restrict__ whhT, const float* __restrict__ bih,
    const float* __restrict__ bhh, float* __restrict__ out)
{
    extern __shared__ char smem[];
    _Float16* xh = (_Float16*)smem;               // phase1: x tile (48 rows); phase2: h16 (32 rows)
    _Float16* gx = (_Float16*)(smem + XH_BYTES);  // 40 rows x 2304 f16 (+b_ih folded)

    const int tid   = threadIdx.x;
    const int lane  = tid & 31;
    const int wave  = tid >> 5;
    const int lhalf = lane & 15;
    const int lgrp  = lane >> 4;
    const int b     = blockIdx.x >> 6;            // 64 row-tiles of 32 per batch
    const int t0    = (blockIdx.x & 63) << 5;
    const int wcol  = wave * 48;                  // each wave owns 48 d-columns of every gate

    // ---- fill x tile (xpad rows [t0, t0+48)) as f16, zero-padded ----
    for (int e = tid; e < XH_ROWS * Dd; e += 256) {
        int r = e / Dd, c = e - r * Dd;
        int xp = t0 + r;                           // xpad row index
        float v = 0.0f;
        if (xp >= (KW - 1) && xp < Ll + (KW - 1))
            v = __builtin_nontemporal_load(&x[((size_t)b * Ll + (xp - (KW - 1))) * Dd + c]);
        xh[r * XH_PITCH + c] = (_Float16)v;
    }
    __syncthreads();

    // per-thread bias values for the 9 (gate, n-tile) columns this thread touches
    float bihv[3][3], bhhv[3][3];
#pragma unroll
    for (int g = 0; g < 3; ++g)
#pragma unroll
        for (int i = 0; i < 3; ++i) {
            int col = g * Dd + wcol + 16 * i + lhalf;
            bihv[g][i] = bih[col];
            bhhv[g][i] = bhh[col];
        }

    // ---- Phase 1: gx = x_tile @ w_ih + b_ih  (three 16-row M blocks) ----
    for (int mb = 0; mb < 3; ++mb) {
        size_t woff = 0;
        asm volatile("" : "+s"(woff));   // opaque offset: no hoist/spill across mb,
        const _Float16* wl = wihT + woff; // but base stays global address space
        v8f acc[3][3];
        const v8f vz = {0.f, 0.f, 0.f, 0.f, 0.f, 0.f, 0.f, 0.f};
#pragma unroll
        for (int g = 0; g < 3; ++g)
#pragma unroll
            for (int i = 0; i < 3; ++i) acc[g][i] = vz;

        for (int kb = 0; kb < Dd / 32; ++kb) {
            const int k0 = kb * 32;
            Frag a;
            const f4* ap = (const f4*)(xh + (lhalf + 16 * mb) * XH_PITCH + k0 + 8 * lgrp);
            a.q[0] = ap[0];
            a.q[1] = ap[2];
#pragma unroll
            for (int g = 0; g < 3; ++g)
#pragma unroll
                for (int i = 0; i < 3; ++i) {
                    Frag w;
                    const f4* bp = (const f4*)(wl +
                        (size_t)(g * Dd + wcol + 16 * i + lhalf) * Dd + k0 + 8 * lgrp);
                    w.q[0] = bp[0];
                    w.q[1] = bp[2];
                    acc[g][i] = __builtin_amdgcn_wmma_f32_16x16x32_f16(
                        false, a.v, false, w.v, (short)0, acc[g][i], false, false);
                }
        }
#pragma unroll
        for (int g = 0; g < 3; ++g)
#pragma unroll
            for (int i = 0; i < 3; ++i)
#pragma unroll
                for (int v = 0; v < 8; ++v) {
                    int m = v + 8 * lgrp + 16 * mb;   // C layout: M = v + 8*(lane/16)
                    if (m < GX_ROWS) {
                        int col = g * Dd + wcol + 16 * i + lhalf;
                        gx[m * GX_PITCH + col] = (_Float16)(acc[g][i][v] + bihv[g][i]);
                    }
                }
    }
    __syncthreads();   // gx visible; all reads of x tile done

    // ---- j = 0: h_prev = 0 -> gh = b_hh, closed form ----
    float h[2][3][8];
#pragma unroll
    for (int mb = 0; mb < 2; ++mb)
#pragma unroll
        for (int i = 0; i < 3; ++i)
#pragma unroll
            for (int v = 0; v < 8; ++v) {
                int m = v + 8 * lgrp + 16 * mb;
                int d = wcol + 16 * i + lhalf;
                float xr = (float)gx[m * GX_PITCH + d];
                float xz = (float)gx[m * GX_PITCH + Dd + d];
                float xn = (float)gx[m * GX_PITCH + 2 * Dd + d];
                float r  = sigf(xr + bhhv[0][i]);
                float z  = sigf(xz + bhhv[1][i]);
                float nn = tanh_fast(xn + r * bhhv[2][i]);
                h[mb][i][v] = (1.0f - z) * nn;
                xh[m * XH_PITCH + d] = (_Float16)h[mb][i][v];   // h16 for next step's A
            }
    __syncthreads();

    // ---- j = 1..7: gh = h @ w_hh, gate update ----
    for (int j = 1; j < KW; ++j) {
        size_t woff = 0;
        asm volatile("" : "+s"(woff));   // opaque offset: no hoist/spill across j
        const _Float16* wh = whhT + woff;
        v8f acc[3][3][2];
        const v8f vz = {0.f, 0.f, 0.f, 0.f, 0.f, 0.f, 0.f, 0.f};
#pragma unroll
        for (int g = 0; g < 3; ++g)
#pragma unroll
            for (int i = 0; i < 3; ++i) { acc[g][i][0] = vz; acc[g][i][1] = vz; }

        for (int kb = 0; kb < Dd / 32; ++kb) {
            const int k0 = kb * 32;
            Frag a0, a1;
            const f4* ap0 = (const f4*)(xh + lhalf * XH_PITCH + k0 + 8 * lgrp);
            const f4* ap1 = (const f4*)(xh + (lhalf + 16) * XH_PITCH + k0 + 8 * lgrp);
            a0.q[0] = ap0[0]; a0.q[1] = ap0[2];
            a1.q[0] = ap1[0]; a1.q[1] = ap1[2];
#pragma unroll
            for (int g = 0; g < 3; ++g)
#pragma unroll
                for (int i = 0; i < 3; ++i) {
                    Frag w;
                    const f4* bp = (const f4*)(wh +
                        (size_t)(g * Dd + wcol + 16 * i + lhalf) * Dd + k0 + 8 * lgrp);
                    w.q[0] = bp[0];
                    w.q[1] = bp[2];
                    acc[g][i][0] = __builtin_amdgcn_wmma_f32_16x16x32_f16(
                        false, a0.v, false, w.v, (short)0, acc[g][i][0], false, false);
                    acc[g][i][1] = __builtin_amdgcn_wmma_f32_16x16x32_f16(
                        false, a1.v, false, w.v, (short)0, acc[g][i][1], false, false);
                }
        }
        __syncthreads();   // all waves done reading h16 before overwrite

#pragma unroll
        for (int mb = 0; mb < 2; ++mb)
#pragma unroll
            for (int i = 0; i < 3; ++i)
#pragma unroll
                for (int v = 0; v < 8; ++v) {
                    int m = v + 8 * lgrp + 16 * mb;
                    int d = wcol + 16 * i + lhalf;
                    int row = m + j;                 // gx row for this step
                    float xr = (float)gx[row * GX_PITCH + d];
                    float xz = (float)gx[row * GX_PITCH + Dd + d];
                    float xn = (float)gx[row * GX_PITCH + 2 * Dd + d];
                    float r  = sigf(xr + acc[0][i][mb][v] + bhhv[0][i]);
                    float z  = sigf(xz + acc[1][i][mb][v] + bhhv[1][i]);
                    float nn = tanh_fast(xn + r * (acc[2][i][mb][v] + bhhv[2][i]));
                    h[mb][i][v] = (1.0f - z) * nn + z * h[mb][i][v];
                    if (j < KW - 1) xh[m * XH_PITCH + d] = (_Float16)h[mb][i][v];
                }
        __syncthreads();
    }

    // ---- write final hidden states [B, L, D] f32 (non-temporal: streaming) ----
#pragma unroll
    for (int mb = 0; mb < 2; ++mb)
#pragma unroll
        for (int i = 0; i < 3; ++i)
#pragma unroll
            for (int v = 0; v < 8; ++v) {
                int m = v + 8 * lgrp + 16 * mb;
                int d = wcol + 16 * i + lhalf;
                __builtin_nontemporal_store(h[mb][i][v],
                    &out[((size_t)b * Ll + t0 + m) * Dd + d]);
            }
}

extern "C" void kernel_launch(void* const* d_in, const int* in_sizes, int n_in,
                              void* d_out, int out_size, void* d_ws, size_t ws_size,
                              hipStream_t stream) {
    (void)in_sizes; (void)n_in; (void)out_size; (void)ws_size;
    const float* x   = (const float*)d_in[0];
    const float* wih = (const float*)d_in[1];
    const float* whh = (const float*)d_in[2];
    const float* bih = (const float*)d_in[3];
    const float* bhh = (const float*)d_in[4];
    float* out = (float*)d_out;

    _Float16* wihT = (_Float16*)d_ws;                 // 2304*768 f16 = 3.54 MB
    _Float16* whhT = wihT + (size_t)Dd * N3;          // 3.54 MB more

    prep_weights<<<(Dd * N3 + 255) / 256, 256, 0, stream>>>(wih, whh, wihT, whhT);

    (void)hipFuncSetAttribute((const void*)localrnn_kernel,
                              hipFuncAttributeMaxDynamicSharedMemorySize, LDS_BYTES);
    localrnn_kernel<<<Bb * (Ll / MT), 256, LDS_BYTES, stream>>>(x, wihT, whhT, bih, bhh, out);
}
